// Decoder_82583631167570
// MI455X (gfx1250) — matrix operand order repaired and checked by reading, not compiled, not run on hardware
//
#include <hip/hip_runtime.h>
#include <hip/hip_bf16.h>

// Problem constants (from reference): V=32000, E=512, H=1024, L=2, T=64, B=32, S=128
typedef __bf16 bf16_t;
typedef bf16_t v16bf __attribute__((ext_vector_type(16)));
typedef bf16_t v8bf  __attribute__((ext_vector_type(8)));
typedef float  v8f   __attribute__((ext_vector_type(8)));

#define KSPLIT 4   // K-slices for the two LSTM-gate GEMMs

// Native gfx1250 TRANS tanh if the toolchain exposes it (V_TANH_F32), else libm.
__device__ __forceinline__ float fast_tanhf(float x) {
#if defined(__has_builtin)
#if __has_builtin(__builtin_amdgcn_tanhf)
  return __builtin_amdgcn_tanhf(x);
#else
  return tanhf(x);
#endif
#else
  return tanhf(x);
#endif
}
__device__ __forceinline__ float sigmoidf(float x) { return 1.f / (1.f + __expf(-x)); }

// ---------------------------------------------------------------------------
// WMMA fragment loaders (CDNA5 16x16x32 bf16 layouts, CDNA5 ISA)
// A (16x32, row-major src): lane<16 -> M=lane, K elems {0..7,16..23};
//                           lane>=16 -> M=lane-16, K elems {8..15,24..31}
// B (32x16) from row-major W[N][K] (B col-major == W rows):
//   lane<16 -> N=lane, K {0..7,16..23}; lane>=16 -> N=lane-16, K {8..15,24..31}
// ---------------------------------------------------------------------------
__device__ __forceinline__ v16bf load_a_frag(const bf16_t* p) {
  v8bf lo = *(const v8bf*)(p);
  v8bf hi = *(const v8bf*)(p + 16);
  v16bf r;
#pragma unroll
  for (int i = 0; i < 8; ++i) { r[i] = lo[i]; r[i + 8] = hi[i]; }
  return r;
}
__device__ __forceinline__ v16bf load_b_frag(const bf16_t* p) {
  v8bf lo = *(const v8bf*)(p);
  v8bf hi = *(const v8bf*)(p + 16);
  v16bf r;
#pragma unroll
  for (int i = 0; i < 8; ++i) { r[i] = lo[i]; r[i + 8] = hi[i]; }
  return r;
}

#define WMMA4(AV, B0, B1, B2, B3)                                               \
  acc0 = __builtin_amdgcn_wmma_f32_16x16x32_bf16(false, AV, false, B0,          \
                                                 (short)0, acc0, false, false); \
  acc1 = __builtin_amdgcn_wmma_f32_16x16x32_bf16(false, AV, false, B1,          \
                                                 (short)0, acc1, false, false); \
  acc2 = __builtin_amdgcn_wmma_f32_16x16x32_bf16(false, AV, false, B2,          \
                                                 (short)0, acc2, false, false); \
  acc3 = __builtin_amdgcn_wmma_f32_16x16x32_bf16(false, AV, false, B3,          \
                                                 (short)0, acc3, false, false)

// C = A[M,K](bf16) @ W[N,K]^T (bf16) [+ bias], C f32 [M,N].
// One wave -> 16x64 strip (A fragment reused across 4 WMMAs).
// NSTEPS = compile-time K-steps (of 32) per K-slice; gridDim.y = #slices and
// slice s covers K range [s*NSTEPS*32, (s+1)*NSTEPS*32), writing C + s*M*ldc.
// Fully unrolled -> pure SSA straight-line code: no loop phis, no rotation
// copies; the scheduler freely hoists fragment loads across WMMA groups.
template <int NSTEPS>
__global__ __launch_bounds__(256) void wmma_gemm_bf16(
    const bf16_t* __restrict__ A, int lda,
    const bf16_t* __restrict__ W, int ldw,
    const float* __restrict__ bias,
    float* __restrict__ C, int ldc,
    int M, int N)
{
  const int lane = threadIdx.x & 31;
  const int wave = blockIdx.x * (blockDim.x >> 5) + (threadIdx.x >> 5);
  const int ncols = N >> 6;
  if (wave >= (M >> 4) * ncols) return;   // wave-uniform: EXEC stays all-ones
  const int tm  = wave / ncols;
  const int tn4 = (wave % ncols) << 2;
  const int l15 = lane & 15;
  const bool hi = lane >= 16;

  const int k0beg = (int)blockIdx.y * (NSTEPS * 32);
  float* Cs = C + (size_t)blockIdx.y * (size_t)M * ldc;

  const bf16_t* arow = A + (size_t)(tm * 16 + l15) * lda + (hi ? 8 : 0) + k0beg;
  const bf16_t* wr0 = W + (size_t)((tn4 + 0) * 16 + l15) * ldw + (hi ? 8 : 0) + k0beg;
  const bf16_t* wr1 = wr0 + (size_t)16 * ldw;
  const bf16_t* wr2 = wr1 + (size_t)16 * ldw;
  const bf16_t* wr3 = wr2 + (size_t)16 * ldw;

  v8f acc0 = {}, acc1 = {}, acc2 = {}, acc3 = {};

  // prologue: load step 0
  v16bf a0  = load_a_frag(arow);
  v16bf b00 = load_b_frag(wr0);
  v16bf b01 = load_b_frag(wr1);
  v16bf b02 = load_b_frag(wr2);
  v16bf b03 = load_b_frag(wr3);

  constexpr int PAIRS = (NSTEPS - 1) / 2;   // pipelined pairs after prologue
#pragma unroll
  for (int pr = 0; pr < PAIRS; ++pr) {
    const int k = 32 + pr * 64;
    // stage 1: load set B at k, compute set A
    v16bf a1  = load_a_frag(arow + k);
    v16bf b10 = load_b_frag(wr0 + k);
    v16bf b11 = load_b_frag(wr1 + k);
    v16bf b12 = load_b_frag(wr2 + k);
    v16bf b13 = load_b_frag(wr3 + k);
    __builtin_prefetch(wr0 + k + 64, 0, 1);   // global_prefetch_b8 -> warm L2
    __builtin_prefetch(wr2 + k + 64, 0, 1);
    WMMA4(a0, b00, b01, b02, b03);
    // stage 2: load set A at k+32, compute set B
    a0  = load_a_frag(arow + k + 32);
    b00 = load_b_frag(wr0 + k + 32);
    b01 = load_b_frag(wr1 + k + 32);
    b02 = load_b_frag(wr2 + k + 32);
    b03 = load_b_frag(wr3 + k + 32);
    WMMA4(a1, b10, b11, b12, b13);
  }
  if constexpr (((NSTEPS - 1) & 1) == 1) {
    // one step left to load: last step index = NSTEPS-1
    const int k = 32 + PAIRS * 64;
    v16bf a1  = load_a_frag(arow + k);
    v16bf b10 = load_b_frag(wr0 + k);
    v16bf b11 = load_b_frag(wr1 + k);
    v16bf b12 = load_b_frag(wr2 + k);
    v16bf b13 = load_b_frag(wr3 + k);
    WMMA4(a0, b00, b01, b02, b03);
    WMMA4(a1, b10, b11, b12, b13);
  } else {
    WMMA4(a0, b00, b01, b02, b03);
  }

  // C/D layout: vec elem r, lane l -> row tm*16 + r + (hi?8:0), col = l&15
  const int m0 = tm * 16 + (hi ? 8 : 0);
  const bool addb = (bias != nullptr) && (gridDim.y == 1);
  v8f accs[4] = {acc0, acc1, acc2, acc3};
#pragma unroll
  for (int j = 0; j < 4; ++j) {
    const int n = (tn4 + j) * 16 + l15;
    const float bv = addb ? bias[n] : 0.0f;
#pragma unroll
    for (int r = 0; r < 8; ++r)
      Cs[(size_t)(m0 + r) * ldc + n] = accs[j][r] + bv;
  }
}

// ---------------------------------------------------------------------------
// Prep kernels (run once per launch)
// ---------------------------------------------------------------------------
__global__ void f2bf_kernel(const float* __restrict__ in, bf16_t* __restrict__ out, long n) {
  long i = (long)blockIdx.x * 256 + threadIdx.x;
  if (i < n) out[i] = (bf16_t)in[i];
}
// Wcat0[4096][2560] = [Wih0 | Whh0]
__global__ void concat_w0_kernel(const float* __restrict__ Wih, const float* __restrict__ Whh,
                                 bf16_t* __restrict__ out) {
  long i = (long)blockIdx.x * 256 + threadIdx.x;
  if (i >= 4096L * 2560) return;
  int n = (int)(i / 2560), k = (int)(i % 2560);
  float v = (k < 1536) ? Wih[(size_t)n * 1536 + k] : Whh[(size_t)n * 1024 + (k - 1536)];
  out[i] = (bf16_t)v;
}
// Wcat1[4096][2048] = [Wih1 | Whh1]
__global__ void concat_w1_kernel(const float* __restrict__ Wih, const float* __restrict__ Whh,
                                 bf16_t* __restrict__ out) {
  long i = (long)blockIdx.x * 256 + threadIdx.x;
  if (i >= 4096L * 2048) return;
  int n = (int)(i >> 11), k = (int)(i & 2047);
  float v = (k < 1024) ? Wih[(size_t)n * 1024 + k] : Whh[(size_t)n * 1024 + (k - 1024)];
  out[i] = (bf16_t)v;
}
__global__ void prep_bias_kernel(const float* a0, const float* b0,
                                 const float* a1, const float* b1,
                                 float* o0, float* o1) {
  int i = blockIdx.x * 256 + threadIdx.x;
  if (i < 4096)      o0[i] = a0[i] + b0[i];
  else if (i < 8192) { int j = i - 4096; o1[j] = a1[j] + b1[j]; }
}
// s_bf[b][s][h] = bf16(memory_bank[s][b][h])
__global__ void prep_s_kernel(const float* __restrict__ mb, bf16_t* __restrict__ out) {
  long i = (long)blockIdx.x * 256 + threadIdx.x;
  if (i >= 4096L * 1024) return;
  int b = (int)(i >> 17);
  int s = (int)((i >> 10) & 127);
  int h = (int)(i & 1023);
  out[i] = (bf16_t)mb[((size_t)s * 32 + b) * 1024 + h];
}
// emb_bf[t][b][e] = bf16(emb[tokens[t][b]][e])
__global__ void prep_emb_kernel(const int* __restrict__ tokens, const float* __restrict__ emb,
                                bf16_t* __restrict__ out) {
  long i = (long)blockIdx.x * 256 + threadIdx.x;
  if (i >= 64L * 32 * 512) return;
  int e = (int)(i & 511);
  int bt = (int)(i >> 9);
  int b = bt & 31, t = bt >> 5;
  int tok = tokens[t * 32 + b];
  out[i] = (bf16_t)emb[(size_t)tok * 512 + e];
}
__global__ void init_state_kernel(const float* h0, const float* c0,
                                  float* h_f, float* c_f, bf16_t* h_b) {
  int i = blockIdx.x * 256 + threadIdx.x;
  if (i < 2 * 32 * 1024) { h_f[i] = h0[i]; c_f[i] = c0[i]; h_b[i] = (bf16_t)h0[i]; }
}

// ---------------------------------------------------------------------------
// Attention: scores = w2 . tanh(psum + proj_h), masked softmax, ctx = aw @ s,
// then assemble x0 = [emb_t | ctx | h_layer0] in bf16. One block per batch row.
// ---------------------------------------------------------------------------
__global__ __launch_bounds__(256) void attn_kernel(
    const float* __restrict__ psum, const float* __restrict__ projh,
    const float* __restrict__ w2, const int* __restrict__ vlens,
    const float* __restrict__ mem, const bf16_t* __restrict__ emb_t,
    const bf16_t* __restrict__ h_b, bf16_t* __restrict__ x0,
    float* __restrict__ attn_out)
{
  const int b = blockIdx.x;
  const int tid = threadIdx.x;
  const int wv = tid >> 5, lane = tid & 31;
  __shared__ float sc[128];
  __shared__ float red[8];

  const float* ph = projh + (size_t)b * 2048;
#pragma unroll 1
  for (int si = 0; si < 16; ++si) {           // 8 waves x 16 s-positions
    const int s = wv * 16 + si;
    const float* pr = psum + ((size_t)(b * 128 + s)) * 2048;
    float acc = 0.f;
    for (int j = lane; j < 2048; j += 32)
      acc += w2[j] * fast_tanhf(pr[j] + ph[j]);
#pragma unroll
    for (int off = 16; off; off >>= 1) acc += __shfl_down(acc, off);
    if (lane == 0) sc[s] = acc;
  }
  __syncthreads();

  const int vl = vlens[b];
  float sv = -1e30f;
  if (tid < 128) { sv = (tid < vl) ? sc[tid] : -1e9f; sc[tid] = sv; }
  float m = sv;
#pragma unroll
  for (int off = 16; off; off >>= 1) m = fmaxf(m, __shfl_down(m, off));
  if (lane == 0) red[wv] = m;
  __syncthreads();
  if (tid == 0) {
    float mm = red[0];
    for (int i = 1; i < 4; ++i) mm = fmaxf(mm, red[i]);
    red[0] = mm;
  }
  __syncthreads();
  const float gmax = red[0];
  float e = (tid < 128) ? __expf(sc[tid] - gmax) : 0.f;
  float ssum = e;
#pragma unroll
  for (int off = 16; off; off >>= 1) ssum += __shfl_down(ssum, off);
  __syncthreads();
  if (lane == 0) red[wv] = ssum;
  __syncthreads();
  if (tid == 0) {
    float tot = 0.f;
    for (int i = 0; i < 4; ++i) tot += red[i];
    red[0] = 1.f / tot;
  }
  __syncthreads();
  const float inv = red[0];
  if (tid < 128) {
    float a = e * inv;
    sc[tid] = a;
    attn_out[(size_t)b * 128 + tid] = a;     // attns[t][b][s]
  }
  __syncthreads();

  for (int h = tid; h < 1024; h += 256) {    // ctx -> x0[:, 512:1536]
    float acc = 0.f;
#pragma unroll 4
    for (int s = 0; s < 128; ++s)
      acc += sc[s] * mem[((size_t)s * 32 + b) * 1024 + h];
    x0[(size_t)b * 2560 + 512 + h] = (bf16_t)acc;
  }
  for (int e2 = tid; e2 < 512; e2 += 256)
    x0[(size_t)b * 2560 + e2] = emb_t[(size_t)b * 512 + e2];
  for (int h = tid; h < 1024; h += 256)
    x0[(size_t)b * 2560 + 1536 + h] = h_b[(size_t)b * 1024 + h];   // layer-0 h
}

// ---------------------------------------------------------------------------
// LSTM cell updates; gate value = bias + sum of KSPLIT partial GEMM results.
// ---------------------------------------------------------------------------
__device__ __forceinline__ float gate_val(const float* __restrict__ gp,
                                          const float* __restrict__ bias,
                                          int b, int col) {
  float v = bias[col];
#pragma unroll
  for (int s = 0; s < KSPLIT; ++s)
    v += gp[(size_t)s * 32 * 4096 + (size_t)b * 4096 + col];
  return v;
}

__global__ __launch_bounds__(256) void lstm_cell0(
    const float* __restrict__ gparts, const float* __restrict__ bias,
    float* __restrict__ h_f, float* __restrict__ c_f,
    bf16_t* __restrict__ h_b, bf16_t* __restrict__ x1)
{
  int i = blockIdx.x * 256 + threadIdx.x;
  if (i >= 32 * 1024) return;
  int b = i >> 10, h = i & 1023;
  float ig = sigmoidf(gate_val(gparts, bias, b, h));
  float fg = sigmoidf(gate_val(gparts, bias, b, 1024 + h));
  float gg = fast_tanhf(gate_val(gparts, bias, b, 2048 + h));
  float og = sigmoidf(gate_val(gparts, bias, b, 3072 + h));
  float c  = fg * c_f[i] + ig * gg;           // layer 0 state at offset 0
  float hn = og * fast_tanhf(c);
  c_f[i] = c; h_f[i] = hn;
  bf16_t hb = (bf16_t)hn;
  h_b[i] = hb;
  x1[(size_t)b * 2048 + h] = hb;                        // x1 = [h1_new | h_L1_prev]
  x1[(size_t)b * 2048 + 1024 + h] = h_b[32 * 1024 + i]; // layer-1 h (pre-update)
}

__global__ __launch_bounds__(256) void lstm_cell1(
    const float* __restrict__ gparts, const float* __restrict__ bias,
    float* __restrict__ h_f, float* __restrict__ c_f,
    bf16_t* __restrict__ h_b, float* __restrict__ dec_out)
{
  int i = blockIdx.x * 256 + threadIdx.x;
  if (i >= 32 * 1024) return;
  int b = i >> 10, h = i & 1023;
  float ig = sigmoidf(gate_val(gparts, bias, b, h));
  float fg = sigmoidf(gate_val(gparts, bias, b, 1024 + h));
  float gg = fast_tanhf(gate_val(gparts, bias, b, 2048 + h));
  float og = sigmoidf(gate_val(gparts, bias, b, 3072 + h));
  const int off = 32 * 1024 + i;              // layer 1 state
  float c  = fg * c_f[off] + ig * gg;
  float hn = og * fast_tanhf(c);
  c_f[off] = c; h_f[off] = hn;
  h_b[off] = (bf16_t)hn;
  dec_out[i] = hn;                            // dec_outs[t][b][h]
}

__global__ void final_copy_kernel(const float* h_f, const float* c_f,
                                  float* oh, float* oc) {
  int i = blockIdx.x * 256 + threadIdx.x;
  if (i < 2 * 32 * 1024) { oh[i] = h_f[i]; oc[i] = c_f[i]; }
}

// ---------------------------------------------------------------------------
// Host launcher
// ---------------------------------------------------------------------------
extern "C" void kernel_launch(void* const* d_in, const int* in_sizes, int n_in,
                              void* d_out, int out_size, void* d_ws, size_t ws_size,
                              hipStream_t stream) {
  const int*   tokens = (const int*)  d_in[0];
  const float* mem    = (const float*)d_in[1];
  const int*   vlens  = (const int*)  d_in[2];
  const float* h0     = (const float*)d_in[3];
  const float* c0     = (const float*)d_in[4];
  const float* emb    = (const float*)d_in[5];
  const float* W1     = (const float*)d_in[6];
  const float* w2     = (const float*)d_in[7];
  const float* Wih0   = (const float*)d_in[8];
  const float* Whh0   = (const float*)d_in[9];
  const float* bih0   = (const float*)d_in[10];
  const float* bhh0   = (const float*)d_in[11];
  const float* Wih1   = (const float*)d_in[12];
  const float* Whh1   = (const float*)d_in[13];
  const float* bih1   = (const float*)d_in[14];
  const float* bhh1   = (const float*)d_in[15];

  // Workspace carve (256B aligned)
  char* p = (char*)d_ws;
  auto alloc = [&](size_t bytes) -> char* {
    char* r = p; p += (bytes + 255) & ~(size_t)255; return r;
  };
  float*  psum    = (float*) alloc(4096ull * 2048 * 4);   // s @ W1s^T (loop-invariant)
  bf16_t* W1b     = (bf16_t*)alloc(2048ull * 2048 * 2);
  bf16_t* Wc0     = (bf16_t*)alloc(4096ull * 2560 * 2);
  bf16_t* Wc1     = (bf16_t*)alloc(4096ull * 2048 * 2);
  bf16_t* s_bf    = (bf16_t*)alloc(4096ull * 1024 * 2);
  bf16_t* emb_bf  = (bf16_t*)alloc(64ull * 32 * 512 * 2);
  float*  bias0   = (float*) alloc(4096 * 4);
  float*  bias1   = (float*) alloc(4096 * 4);
  float*  h_f     = (float*) alloc(2ull * 32 * 1024 * 4);
  float*  c_f     = (float*) alloc(2ull * 32 * 1024 * 4);
  bf16_t* h_b     = (bf16_t*)alloc(2ull * 32 * 1024 * 2);
  float*  projh   = (float*) alloc(32ull * 2048 * 4);
  bf16_t* x0      = (bf16_t*)alloc(32ull * 2560 * 2);
  bf16_t* x1      = (bf16_t*)alloc(32ull * 2048 * 2);
  float*  gates0p = (float*) alloc((size_t)KSPLIT * 32 * 4096 * 4);
  float*  gates1p = (float*) alloc((size_t)KSPLIT * 32 * 4096 * 4);

  // Output layout: dec_outs[T,B,H] | hT[2,B,H] | cT[2,B,H] | attns[T,B,S]
  float* out      = (float*)d_out;
  float* dec_outs = out;
  float* out_hT   = out + 64 * 32 * 1024;
  float* out_cT   = out_hT + 2 * 32 * 1024;
  float* attns    = out_cT + 2 * 32 * 1024;

  // --- one-time prep ---
  f2bf_kernel<<<(2048 * 2048 + 255) / 256, 256, 0, stream>>>(W1, W1b, 2048L * 2048);
  concat_w0_kernel<<<(int)((4096L * 2560 + 255) / 256), 256, 0, stream>>>(Wih0, Whh0, Wc0);
  concat_w1_kernel<<<(int)((4096L * 2048 + 255) / 256), 256, 0, stream>>>(Wih1, Whh1, Wc1);
  prep_bias_kernel<<<(8192 + 255) / 256, 256, 0, stream>>>(bih0, bhh0, bih1, bhh1, bias0, bias1);
  prep_s_kernel<<<(int)((4096L * 1024 + 255) / 256), 256, 0, stream>>>(mem, s_bf);
  prep_emb_kernel<<<(int)((64L * 32 * 512 + 255) / 256), 256, 0, stream>>>(tokens, emb, emb_bf);
  init_state_kernel<<<(2 * 32 * 1024 + 255) / 256, 256, 0, stream>>>(h0, c0, h_f, c_f, h_b);

  auto gemm_bx = [](int M, int N) { return ((M / 16) * (N / 64) + 7) / 8; };

  // psum[B*S, 2H] = s_bf @ W1[:, :H]^T   (once; K=1024 -> 32 steps, 1 slice)
  wmma_gemm_bf16<32><<<dim3(gemm_bx(4096, 2048), 1), 256, 0, stream>>>(
      s_bf, 1024, W1b, 2048, nullptr, psum, 2048, 4096, 2048);

  // --- recurrence ---
  for (int t = 0; t < 64; ++t) {
    // proj_h[B, 2H] = h_layer1 @ W1[:, H:]^T   (K=1024 -> 32 steps, 1 slice)
    wmma_gemm_bf16<32><<<dim3(gemm_bx(32, 2048), 1), 256, 0, stream>>>(
        h_b + 32 * 1024, 1024, W1b + 1024, 2048, nullptr, projh, 2048, 32, 2048);
    // attention + ctx + x0 assembly
    attn_kernel<<<32, 256, 0, stream>>>(
        psum, projh, w2, vlens, mem, emb_bf + (size_t)t * 32 * 512, h_b, x0,
        attns + (size_t)t * 32 * 128);
    // layer-0 gates: K=2560, 4 slices -> 20 steps each
    wmma_gemm_bf16<20><<<dim3(gemm_bx(32, 4096), KSPLIT), 256, 0, stream>>>(
        x0, 2560, Wc0, 2560, nullptr, gates0p, 4096, 32, 4096);
    lstm_cell0<<<128, 256, 0, stream>>>(gates0p, bias0, h_f, c_f, h_b, x1);
    // layer-1 gates: K=2048, 4 slices -> 16 steps each
    wmma_gemm_bf16<16><<<dim3(gemm_bx(32, 4096), KSPLIT), 256, 0, stream>>>(
        x1, 2048, Wc1, 2048, nullptr, gates1p, 4096, 32, 4096);
    lstm_cell1<<<128, 256, 0, stream>>>(gates1p, bias1, h_f, c_f, h_b,
                                        dec_outs + (size_t)t * 32 * 1024);
  }

  final_copy_kernel<<<(2 * 32 * 1024 + 255) / 256, 256, 0, stream>>>(h_f, c_f, out_hT, out_cT);
}
